// CFConv_34093450396365
// MI455X (gfx1250) — compile-verified, matrix operand order checked by reading, not run on hardware
//
#include <hip/hip_runtime.h>
#include <hip/hip_bf16.h>

typedef __attribute__((ext_vector_type(16))) __bf16 v16bf;
typedef __attribute__((ext_vector_type(8)))  __bf16 v8bf;
typedef __attribute__((ext_vector_type(8)))  float  v8f;
typedef __attribute__((ext_vector_type(4)))  unsigned int v4u;
typedef __attribute__((ext_vector_type(8)))  int v8i;
typedef __attribute__((ext_vector_type(4)))  int v4i;

#define RBF_DIM 128
#define DIM 64
#define WAVES_PER_BLOCK 8

// LDS strides, all 16B-aligned, chosen so the 16 active lanes of a half-wave
// hit distinct bank groups.
#define W1T_STRIDE 136    // bf16 elems: 272 B per column of W1^T
#define W2T_STRIDE 88     // bf16 elems: 176 B per column of W2^T
#define H_STRIDE   88     // bf16 elems: 176 B per row of per-wave H tile
#define RBF_STRIDE 132    // fp32 elems: 528 B per row of the TDM-staged tile
                          // (= 128 data DWORDs + 4 pad DWORDs from the TDM)

__device__ __forceinline__ v16bf load_b_frag(const __bf16* p) {
    v8bf lo = *(const v8bf*)(p);
    v8bf hi = *(const v8bf*)(p + 16);
    v16bf b;
#pragma unroll
    for (int i = 0; i < 8; ++i) { b[i] = lo[i]; b[8 + i] = hi[i]; }
    return b;
}

__device__ __forceinline__ float shifted_softplus(float v) {
    float t = 0.5f * v;
    return (t > 14.0f) ? v : 2.0f * __logf(1.0f + __expf(t));
}

__device__ __forceinline__ unsigned int rfl(unsigned int v) {
    return (unsigned int)__builtin_amdgcn_readfirstlane((int)v);
}

__global__ __launch_bounds__(256) void cfconv_fused_kernel(
    const float* __restrict__ rbf, const float* __restrict__ x,
    const int* __restrict__ src, const int* __restrict__ dst,
    const float* __restrict__ W1, const float* __restrict__ b1,
    const float* __restrict__ W2, const float* __restrict__ b2,
    float* __restrict__ out, int nEdges, int nTiles)
{
    __shared__ __bf16 sW1T[DIM * W1T_STRIDE];                 // W1^T [col][k]
    __shared__ __bf16 sW2T[DIM * W2T_STRIDE];                 // W2^T [col][k]
    __shared__ __bf16 sH[WAVES_PER_BLOCK * 16 * H_STRIDE];    // per-wave H
    __shared__ float  sRBF[WAVES_PER_BLOCK * 16 * RBF_STRIDE];// per-wave rbf tile

    const int tid = threadIdx.x;

    // Stage W1^T and W2^T as bf16 into LDS (coalesced global reads).
    for (int i = tid; i < RBF_DIM * DIM; i += blockDim.x) {
        int k = i >> 6, c = i & 63;
        sW1T[c * W1T_STRIDE + k] = (__bf16)W1[i];
    }
    for (int i = tid; i < DIM * DIM; i += blockDim.x) {
        int k = i >> 6, c = i & 63;
        sW2T[c * W2T_STRIDE + k] = (__bf16)W2[i];
    }
    __syncthreads();

    const int wave = tid >> 5;
    const int lane = tid & 31;
    const int hx   = lane >> 4;   // half-wave select (K-group / M-group)
    const int m    = lane & 15;
    __bf16* myH   = &sH[wave * 16 * H_STRIDE];
    float*  myRBF = &sRBF[wave * 16 * RBF_STRIDE];

    const int tile = blockIdx.x * WAVES_PER_BLOCK + wave;
    if (tile >= nTiles) return;
    const int eBase = tile * 16;

    // -------- TDM: DMA the 16x128 fp32 rbf tile into LDS with row padding ----
    {
        int rowsLeft = nEdges - eBase;
        if (rowsLeft > 16) rowsLeft = 16;
        const unsigned int lds_off = (unsigned int)(uintptr_t)myRBF;
        const unsigned long long ga =
            (unsigned long long)(uintptr_t)(rbf + (size_t)eBase * RBF_DIM);

        // D# group 0: count=1 | lds_addr | global_addr | type=2
        v4u g0;
        g0[0] = rfl(1u);
        g0[1] = rfl(lds_off);
        g0[2] = rfl((unsigned int)ga);
        g0[3] = rfl((unsigned int)((ga >> 32) & 0x01FFFFFFu) | (2u << 30));

        // D# group 1: data_size=4B, pad 4 DWORDs every 128 DWORDs,
        // tensor_dim0=128, tensor_dim1=rowsLeft (OOB rows read as zero),
        // tile_dim0=128, tile_dim1=rowsLeft, tensor_dim0_stride=128.
        v8i g1;
        g1[0] = (int)rfl((2u << 16) | (1u << 20) | (6u << 22) | (3u << 25));
        g1[1] = (int)rfl(((unsigned int)RBF_DIM & 0xFFFFu) << 16);
        g1[2] = (int)rfl(((unsigned int)rowsLeft & 0xFFFFu) << 16);
        g1[3] = (int)rfl((unsigned int)RBF_DIM << 16);
        g1[4] = (int)rfl((unsigned int)rowsLeft & 0xFFFFu);
        g1[5] = (int)rfl((unsigned int)RBF_DIM);
        g1[6] = 0;
        g1[7] = 0;

        v4i g2 = {0, 0, 0, 0};
        v4i g3 = {0, 0, 0, 0};
        v8i g4 = {0, 0, 0, 0, 0, 0, 0, 0};
        __builtin_amdgcn_tensor_load_to_lds(g0, g1, g2, g3, g4, 0);
    }

    // ---------------- GEMM1: H1 = rbf_tile(16x128) @ W1(128x64) ------------
    v8f zero = {};
    v8f acc[4];
#pragma unroll
    for (int n = 0; n < 4; ++n) acc[n] = zero;

    asm volatile("s_wait_tensorcnt 0x0" ::: "memory");
    const float* rowP = myRBF + m * RBF_STRIDE;

#pragma unroll
    for (int kk = 0; kk < 4; ++kk) {
        const int k0 = kk * 32 + hx * 8;
        // A-fragment (16-bit A layout): lane holds K = {k0..k0+7, k0+16..k0+23}
        const float4 f0 = *(const float4*)(rowP + k0);
        const float4 f1 = *(const float4*)(rowP + k0 + 4);
        const float4 g0v = *(const float4*)(rowP + k0 + 16);
        const float4 g1v = *(const float4*)(rowP + k0 + 20);
        v16bf a;
        a[0]  = (__bf16)f0.x;  a[1]  = (__bf16)f0.y;  a[2]  = (__bf16)f0.z;  a[3]  = (__bf16)f0.w;
        a[4]  = (__bf16)f1.x;  a[5]  = (__bf16)f1.y;  a[6]  = (__bf16)f1.z;  a[7]  = (__bf16)f1.w;
        a[8]  = (__bf16)g0v.x; a[9]  = (__bf16)g0v.y; a[10] = (__bf16)g0v.z; a[11] = (__bf16)g0v.w;
        a[12] = (__bf16)g1v.x; a[13] = (__bf16)g1v.y; a[14] = (__bf16)g1v.z; a[15] = (__bf16)g1v.w;

#pragma unroll
        for (int n = 0; n < 4; ++n) {
            const int col = n * 16 + m;
            v16bf b = load_b_frag(&sW1T[col * W1T_STRIDE + k0]);
            acc[n] = __builtin_amdgcn_wmma_f32_16x16x32_bf16(
                false, a, false, b, (short)0, acc[n], false, false);
        }
    }

    // Bias + shifted softplus, stash H as bf16 in this wave's LDS slice.
    // C layout: vgpr r, lanes 0-15 -> (M=r, N=m); lanes 16-31 -> (M=r+8, N=m)
#pragma unroll
    for (int n = 0; n < 4; ++n) {
        const int col = n * 16 + m;
        const float bias = b1[col];
#pragma unroll
        for (int r = 0; r < 8; ++r) {
            const int M = r + hx * 8;
            myH[M * H_STRIDE + col] = (__bf16)shifted_softplus(acc[n][r] + bias);
        }
    }
    asm volatile("s_wait_dscnt 0" ::: "memory");

    // ---------------- GEMM2: H2 = H1(16x64) @ W2(64x64) --------------------
    v8f dcc[4];
#pragma unroll
    for (int n = 0; n < 4; ++n) dcc[n] = zero;

#pragma unroll
    for (int kk = 0; kk < 2; ++kk) {
        const int k0 = kk * 32 + hx * 8;
        v16bf a2 = load_b_frag(&myH[m * H_STRIDE + k0]);
#pragma unroll
        for (int n = 0; n < 4; ++n) {
            const int col = n * 16 + m;
            v16bf b = load_b_frag(&sW2T[col * W2T_STRIDE + k0]);
            dcc[n] = __builtin_amdgcn_wmma_f32_16x16x32_bf16(
                false, a2, false, b, (short)0, dcc[n], false, false);
        }
    }

    // ---------------- epilogue: msg = x[src] * h2, scatter-add to out ------
    float bb[4];
#pragma unroll
    for (int n = 0; n < 4; ++n) bb[n] = b2[n * 16 + m];

#pragma unroll
    for (int r = 0; r < 8; ++r) {
        const int e = eBase + hx * 8 + r;
        if (e < nEdges) {
            const int s = src[e];
            const int d = dst[e];
            const float* xrow = x + (size_t)s * DIM;
            float* orow = out + (size_t)d * DIM;
#pragma unroll
            for (int n = 0; n < 4; ++n) {
                const int col = n * 16 + m;
                const float h = dcc[n][r] + bb[n];
                atomicAdd(orow + col, h * xrow[col]);
            }
        }
    }
}

extern "C" void kernel_launch(void* const* d_in, const int* in_sizes, int n_in,
                              void* d_out, int out_size, void* d_ws, size_t ws_size,
                              hipStream_t stream) {
    const float* rbf = (const float*)d_in[0];
    const float* x   = (const float*)d_in[1];
    const int*   src = (const int*)d_in[2];
    const int*   dst = (const int*)d_in[3];
    const float* W1  = (const float*)d_in[4];
    const float* b1  = (const float*)d_in[5];
    const float* W2  = (const float*)d_in[6];
    const float* b2  = (const float*)d_in[7];
    float* out = (float*)d_out;

    (void)d_ws; (void)ws_size; (void)n_in;

    // out is poisoned by the harness; the scatter-add needs zeros.
    (void)hipMemsetAsync(out, 0, (size_t)out_size * sizeof(float), stream);

    const int nEdges = in_sizes[2];              // src length == N_EDGES
    const int nTiles = (nEdges + 15) / 16;
    const int blocks = (nTiles + WAVES_PER_BLOCK - 1) / WAVES_PER_BLOCK;
    cfconv_fused_kernel<<<blocks, WAVES_PER_BLOCK * 32, 0, stream>>>(
        rbf, x, src, dst, W1, b1, W2, b2, out, nEdges, nTiles);
}
